// FlashMoEModel_34170759807255
// MI455X (gfx1250) — matrix operand order repaired and checked by reading, not compiled, not run on hardware
//
#include <hip/hip_runtime.h>
#include <hip/hip_bf16.h>

// ---------------------------------------------------------------------------
// FlashMoE forward for MI455X (gfx1250, wave32, WMMA bf16 16x16x32).
// encoded = x @ W_enc^T + b ; logits = enc @ Wg^T ; top2 routing;
// s = silu(enc @ U_e^T) * w * gamma ; out = encoded + s @ V_e^T (sparse).
// GEMMs: v_wmma_f32_16x16x32_bf16, fp32 accum, double-buffered LDS tiles
// filled via async global->LDS copies (ASYNCcnt).
// ---------------------------------------------------------------------------

#define DM    2048
#define NEXP  16
#define RLOW  128
#define NTOK  16384

typedef __bf16 bf16_t;
typedef bf16_t bfrag16 __attribute__((ext_vector_type(16)));
typedef bf16_t bfrag8  __attribute__((ext_vector_type(8)));
typedef bf16_t bf16x4  __attribute__((ext_vector_type(4)));
typedef float  f32x8   __attribute__((ext_vector_type(8)));
typedef int    v4i     __attribute__((ext_vector_type(4)));

union FragU { bfrag16 v; bfrag8 h[2]; };

// ---------------------------------------------------------------------------
// Async global->LDS 16B copy (GLOBAL_LOAD_ASYNC_TO_LDS_B128, ASYNCcnt).
// Builtin signature (from hipcc diagnostic): (int4 AS1*, int4 AS3*, imm, imm).
// ---------------------------------------------------------------------------
static __device__ __forceinline__ void async_copy_b128(const bf16_t* g, bf16_t* l) {
#if __has_builtin(__builtin_amdgcn_global_load_async_to_lds_b128)
    __builtin_amdgcn_global_load_async_to_lds_b128(
        (__attribute__((address_space(1))) v4i*)g,
        (__attribute__((address_space(3))) v4i*)l, 0, 0);
#else
    unsigned loff = (unsigned)(unsigned long)(__attribute__((address_space(3))) void*)l;
    asm volatile("global_load_async_to_lds_b128 %0, %1, off"
                 :: "v"(loff), "v"(g) : "memory");
#endif
}

static __device__ __forceinline__ void wait_async0() {
#if __has_builtin(__builtin_amdgcn_s_wait_asynccnt)
    __builtin_amdgcn_s_wait_asynccnt(0);
#else
    asm volatile("s_wait_asynccnt 0x0" ::: "memory");
#endif
}

// A-matrix 16x32 bf16 fragment (ISA 7.12.2): lane<16 -> K 0..7 & 16..23,
// lane>=16 -> K 8..15 & 24..31, row M = lane&15.
static __device__ __forceinline__ bfrag16
load_frag_a(const bf16_t* tile, int row0, int ks, int stride, int lane) {
    const int r  = row0 + (lane & 15);
    const int kb = ks + ((lane >> 4) << 3);
    const bf16_t* p = tile + r * stride + kb;
    FragU u;
    u.h[0] = *(const bfrag8*)(p);
    u.h[1] = *(const bfrag8*)(p + 16);
    return u.v;
}

// B-matrix 32x16 bf16 fragment: lanes 0..15 hold K=0..7 & 16..23 of column N=lane,
// lanes 16..31 hold K=8..15 & 24..31.
static __device__ __forceinline__ bfrag16
load_frag_b(const bf16_t* tile, int col0, int ks, int stride, int lane) {
    const int c  = col0 + (lane & 15);
    const int kb = ks + ((lane >> 4) << 3);
    const bf16_t* p = tile + c * stride + kb;
    FragU u;
    u.h[0] = *(const bfrag8*)(p);
    u.h[1] = *(const bfrag8*)(p + 16);
    return u.v;
}

// ---------------------------------------------------------------------------
__global__ void cvt_f32_bf16(const float* __restrict__ in,
                             bf16_t* __restrict__ out, size_t n4) {
    // n4 = number of float4 groups (all our sizes are multiples of 4)
    size_t i = (size_t)blockIdx.x * blockDim.x + threadIdx.x;
    size_t stride = (size_t)gridDim.x * blockDim.x;
    for (; i < n4; i += stride) {
        float4 v = ((const float4*)in)[i];
        bf16x4 o;
        o.x = (bf16_t)v.x; o.y = (bf16_t)v.y;
        o.z = (bf16_t)v.z; o.w = (bf16_t)v.w;
        ((bf16x4*)out)[i] = o;
    }
}

__global__ void zero_counts(int* counts, int* cursors) {
    int t = threadIdx.x;
    if (t < NEXP) { counts[t] = 0; cursors[t] = 0; }
}

__global__ void scan_offsets(const int* __restrict__ counts, int* __restrict__ offsets) {
    if (threadIdx.x == 0) {
        int a = 0;
        for (int m = 0; m < NEXP; ++m) { offsets[m] = a; a += counts[m]; }
        offsets[NEXP] = a;
    }
}

__global__ void fill_perm(const int* __restrict__ toke, const int* __restrict__ offsets,
                          int* __restrict__ cursors, int* __restrict__ perm) {
    int b = blockIdx.x * blockDim.x + threadIdx.x;
    if (b >= NTOK) return;
    for (int s = 0; s < 2; ++s) {
        int e = toke[2 * b + s];
        int p = atomicAdd(&cursors[e], 1);
        perm[offsets[e] + p] = 2 * b + s;
    }
}

// ---------------------------------------------------------------------------
// encoded = x @ W_enc^T + bias. 128x128 block tile, K-step 64, 8 waves of
// 32x64 (2x4 wmma tiles). Double-buffered async tile pipeline.
// ---------------------------------------------------------------------------
__global__ __launch_bounds__(256) void gemm_encode(
    const bf16_t* __restrict__ X, const bf16_t* __restrict__ W,
    const float* __restrict__ bias,
    float* __restrict__ encf, bf16_t* __restrict__ encb) {
    constexpr int BM = 128, BK = 64, SK = BK + 8, NIT = DM / BK;
    __shared__ bf16_t As[2][BM * SK];
    __shared__ bf16_t Bs[2][BM * SK];
    const int tid = threadIdx.x, lane = tid & 31, wave = tid >> 5;
    const int wm = (wave >> 1) * 32, wn = (wave & 1) * 64;
    const int m0 = blockIdx.x * BM, n0 = blockIdx.y * BM;

    f32x8 acc[2][4] = {};

    const int lr = tid >> 1, lh = tid & 1;
    const bf16_t* gA = X + (size_t)(m0 + lr) * DM + lh * 32;
    const bf16_t* gB = W + (size_t)(n0 + lr) * DM + lh * 32;
    const int soff = lr * SK + lh * 32;

    auto issue = [&](int k0, int buf) {
#pragma unroll
        for (int i = 0; i < 4; ++i) {
            async_copy_b128(gA + k0 + i * 8, &As[buf][soff + i * 8]);
            async_copy_b128(gB + k0 + i * 8, &Bs[buf][soff + i * 8]);
        }
    };

    issue(0, 0);
    for (int it = 0; it < NIT; ++it) {
        const int buf = it & 1;
        wait_async0();
        __syncthreads();
        if (it + 1 < NIT) issue((it + 1) * BK, buf ^ 1);
        const bf16_t* At = As[buf];
        const bf16_t* Bt = Bs[buf];
#pragma unroll
        for (int ks = 0; ks < BK; ks += 32) {
            bfrag16 af[2], bfv[4];
#pragma unroll
            for (int i = 0; i < 2; ++i) af[i] = load_frag_a(At, wm + 16 * i, ks, SK, lane);
#pragma unroll
            for (int j = 0; j < 4; ++j) bfv[j] = load_frag_b(Bt, wn + 16 * j, ks, SK, lane);
#pragma unroll
            for (int i = 0; i < 2; ++i)
#pragma unroll
                for (int j = 0; j < 4; ++j)
                    acc[i][j] = __builtin_amdgcn_wmma_f32_16x16x32_bf16(
                        false, af[i], false, bfv[j], (short)0, acc[i][j], false, false);
        }
    }

#pragma unroll
    for (int i = 0; i < 2; ++i) {
        const int rb = m0 + wm + 16 * i + ((lane >> 4) << 3);
#pragma unroll
        for (int j = 0; j < 4; ++j) {
            const int c = n0 + wn + 16 * j + (lane & 15);
            const float bv = bias[c];
#pragma unroll
            for (int g = 0; g < 8; ++g) {
                float v = acc[i][j][g] + bv;
                size_t idx = (size_t)(rb + g) * DM + c;
                encf[idx] = v;
                encb[idx] = (bf16_t)v;
            }
        }
    }
}

// ---------------------------------------------------------------------------
// Gating GEMM (N=16) fused with top-2 routing, softmax, gamma-fold, counts.
// ---------------------------------------------------------------------------
__global__ __launch_bounds__(256) void router_kernel(
    const bf16_t* __restrict__ enc, const bf16_t* __restrict__ Wg,
    const float* __restrict__ gamma,
    float* __restrict__ wts, int* __restrict__ toke, int* __restrict__ counts) {
    __shared__ float lg[8][16][17];
    const int tid = threadIdx.x, lane = tid & 31, wave = tid >> 5;
    const int m0 = blockIdx.x * 128 + wave * 16;
    const int ar = m0 + (lane & 15);
    const int br = lane & 15;
    const int ka = (lane >> 4) << 3;
    const int kb = (lane >> 4) << 3;

    f32x8 acc = {};
    for (int k0 = 0; k0 < DM; k0 += 32) {
        FragU a, b;
        const bf16_t* pa = enc + (size_t)ar * DM + k0 + ka;
        a.h[0] = *(const bfrag8*)pa;
        a.h[1] = *(const bfrag8*)(pa + 16);
        const bf16_t* pb = Wg + (size_t)br * DM + k0 + kb;
        b.h[0] = *(const bfrag8*)pb;
        b.h[1] = *(const bfrag8*)(pb + 16);
        acc = __builtin_amdgcn_wmma_f32_16x16x32_bf16(
            false, a.v, false, b.v, (short)0, acc, false, false);
    }
#pragma unroll
    for (int g = 0; g < 8; ++g)
        lg[wave][g + ((lane >> 4) << 3)][lane & 15] = acc[g];  // logits, TAU==1
    __syncthreads();

    if (lane < 16) {
        const int row = lane;
        float v[16];
#pragma unroll
        for (int m = 0; m < 16; ++m) v[m] = lg[wave][row][m];
        int i1 = 0; float v1 = v[0];
#pragma unroll
        for (int m = 1; m < 16; ++m) if (v[m] > v1) { v1 = v[m]; i1 = m; }
        int i2 = (i1 == 0) ? 1 : 0; float v2 = v[i2];
#pragma unroll
        for (int m = 0; m < 16; ++m)
            if (m != i1 && v[m] > v2) { v2 = v[m]; i2 = m; }
        // softmax over [v1, v2] (v1 is the max): e0=1, e1=exp(v2-v1)
        float e1  = __expf(v2 - v1);
        float inv = 1.0f / (1.0f + e1 + 1e-12f);
        float w0 = inv, w1 = e1 * inv;
        if (!(w0 > 1e-12f)) w0 = 0.0f;
        if (!(w1 > 1e-12f)) w1 = 0.0f;
        const int b = m0 + row;
        wts[2 * b]     = w0 * gamma[i1];
        wts[2 * b + 1] = w1 * gamma[i2];
        toke[2 * b]     = i1;
        toke[2 * b + 1] = i2;
        atomicAdd(&counts[i1], 1);
        atomicAdd(&counts[i2], 1);
    }
}

// ---------------------------------------------------------------------------
// Per-expert: s = silu(enc_gathered @ U_e^T) * w, stored bf16 [slot, R].
// ---------------------------------------------------------------------------
__global__ __launch_bounds__(256) void gemm_expert_s(
    const bf16_t* __restrict__ enc, const bf16_t* __restrict__ U,
    const float* __restrict__ wts, const int* __restrict__ perm,
    const int* __restrict__ offsets, const int* __restrict__ counts,
    bf16_t* __restrict__ sg) {
    constexpr int BM = 128, BK = 64, SK = BK + 8, NIT = DM / BK;
    const int e = blockIdx.y;
    const int cnt = counts[e];
    const int m0 = blockIdx.x * BM;
    if (m0 >= cnt) return;
    const int off = offsets[e];

    __shared__ bf16_t As[2][BM * SK];
    __shared__ bf16_t Bs[2][BM * SK];
    __shared__ int   tokS[BM];
    __shared__ float wS[BM];

    const int tid = threadIdx.x, lane = tid & 31, wave = tid >> 5;
    const int wm = (wave >> 1) * 32, wn = (wave & 1) * 64;

    if (tid < BM) {
        int s  = m0 + tid;
        int sc = (s < cnt) ? s : (cnt - 1);
        int pe = perm[off + sc];
        tokS[tid] = pe >> 1;
        wS[tid]   = (s < cnt) ? wts[pe] : 0.0f;
    }
    __syncthreads();

    f32x8 acc[2][4] = {};
    const bf16_t* Ue = U + (size_t)e * RLOW * DM;
    const int lr = tid >> 1, lh = tid & 1;
    const int tok = tokS[lr];
    const bf16_t* gA = enc + (size_t)tok * DM + lh * 32;
    const bf16_t* gB = Ue + (size_t)lr * DM + lh * 32;
    const int soff = lr * SK + lh * 32;

    auto issue = [&](int k0, int buf) {
#pragma unroll
        for (int i = 0; i < 4; ++i) {
            async_copy_b128(gA + k0 + i * 8, &As[buf][soff + i * 8]);
            async_copy_b128(gB + k0 + i * 8, &Bs[buf][soff + i * 8]);
        }
    };

    issue(0, 0);
    for (int it = 0; it < NIT; ++it) {
        const int buf = it & 1;
        wait_async0();
        __syncthreads();
        if (it + 1 < NIT) issue((it + 1) * BK, buf ^ 1);
        const bf16_t* At = As[buf];
        const bf16_t* Bt = Bs[buf];
#pragma unroll
        for (int ks = 0; ks < BK; ks += 32) {
            bfrag16 af[2], bfv[4];
#pragma unroll
            for (int i = 0; i < 2; ++i) af[i] = load_frag_a(At, wm + 16 * i, ks, SK, lane);
#pragma unroll
            for (int j = 0; j < 4; ++j) bfv[j] = load_frag_b(Bt, wn + 16 * j, ks, SK, lane);
#pragma unroll
            for (int i = 0; i < 2; ++i)
#pragma unroll
                for (int j = 0; j < 4; ++j)
                    acc[i][j] = __builtin_amdgcn_wmma_f32_16x16x32_bf16(
                        false, af[i], false, bfv[j], (short)0, acc[i][j], false, false);
        }
    }

#pragma unroll
    for (int i = 0; i < 2; ++i) {
        const int rb = wm + 16 * i + ((lane >> 4) << 3);
#pragma unroll
        for (int j = 0; j < 4; ++j) {
            const int c = wn + 16 * j + (lane & 15);  // r index in [0,128)
#pragma unroll
            for (int g = 0; g < 8; ++g) {
                const int sl = rb + g;
                if (m0 + sl < cnt) {
                    float s = acc[i][j][g];
                    float v = s / (1.0f + __expf(-s)) * wS[sl];  // silu * w*gamma
                    sg[(size_t)(off + m0 + sl) * RLOW + c] = (bf16_t)v;
                }
            }
        }
    }
}

// ---------------------------------------------------------------------------
// Per-expert: delta = s @ V_e^T (K = R = 128), atomically added into out.
// ---------------------------------------------------------------------------
__global__ __launch_bounds__(256) void gemm_expert_delta(
    const bf16_t* __restrict__ sg, const bf16_t* __restrict__ V,
    const int* __restrict__ perm, const int* __restrict__ offsets,
    const int* __restrict__ counts, float* __restrict__ out) {
    constexpr int BM = 128, BK = 64, SK = BK + 8, NIT = RLOW / BK;
    const int e = blockIdx.z;
    const int cnt = counts[e];
    const int m0 = blockIdx.x * BM;
    if (m0 >= cnt) return;
    const int n0 = blockIdx.y * BM;
    const int off = offsets[e];

    __shared__ bf16_t As[2][BM * SK];
    __shared__ bf16_t Bs[2][BM * SK];
    __shared__ int tokS[BM];

    const int tid = threadIdx.x, lane = tid & 31, wave = tid >> 5;
    const int wm = (wave >> 1) * 32, wn = (wave & 1) * 64;

    if (tid < BM) {
        int s  = m0 + tid;
        int sc = (s < cnt) ? s : (cnt - 1);
        tokS[tid] = perm[off + sc] >> 1;
    }
    __syncthreads();

    f32x8 acc[2][4] = {};
    const bf16_t* Ve = V + (size_t)e * DM * RLOW;
    const int lr = tid >> 1, lh = tid & 1;
    const int ars = m0 + lr;
    const int arow = off + ((ars < cnt) ? ars : (cnt - 1));
    const bf16_t* gA = sg + (size_t)arow * RLOW + lh * 32;
    const bf16_t* gB = Ve + (size_t)(n0 + lr) * RLOW + lh * 32;
    const int soff = lr * SK + lh * 32;

    auto issue = [&](int k0, int buf) {
#pragma unroll
        for (int i = 0; i < 4; ++i) {
            async_copy_b128(gA + k0 + i * 8, &As[buf][soff + i * 8]);
            async_copy_b128(gB + k0 + i * 8, &Bs[buf][soff + i * 8]);
        }
    };

    issue(0, 0);
    for (int it = 0; it < NIT; ++it) {
        const int buf = it & 1;
        wait_async0();
        __syncthreads();
        if (it + 1 < NIT) issue((it + 1) * BK, buf ^ 1);
        const bf16_t* At = As[buf];
        const bf16_t* Bt = Bs[buf];
#pragma unroll
        for (int ks = 0; ks < BK; ks += 32) {
            bfrag16 af[2], bfv[4];
#pragma unroll
            for (int i = 0; i < 2; ++i) af[i] = load_frag_a(At, wm + 16 * i, ks, SK, lane);
#pragma unroll
            for (int j = 0; j < 4; ++j) bfv[j] = load_frag_b(Bt, wn + 16 * j, ks, SK, lane);
#pragma unroll
            for (int i = 0; i < 2; ++i)
#pragma unroll
                for (int j = 0; j < 4; ++j)
                    acc[i][j] = __builtin_amdgcn_wmma_f32_16x16x32_bf16(
                        false, af[i], false, bfv[j], (short)0, acc[i][j], false, false);
        }
    }

#pragma unroll
    for (int i = 0; i < 2; ++i) {
        const int rb = wm + 16 * i + ((lane >> 4) << 3);
#pragma unroll
        for (int j = 0; j < 4; ++j) {
            const int c = n0 + wn + 16 * j + (lane & 15);
#pragma unroll
            for (int g = 0; g < 8; ++g) {
                const int sl = rb + g;
                if (m0 + sl < cnt) {
                    const int tok = tokS[sl];
                    atomicAdd(&out[(size_t)tok * DM + c], acc[i][j][g]);
                }
            }
        }
    }
}

// ---------------------------------------------------------------------------
extern "C" void kernel_launch(void* const* d_in, const int* in_sizes, int n_in,
                              void* d_out, int out_size, void* d_ws, size_t ws_size,
                              hipStream_t stream) {
    (void)in_sizes; (void)n_in; (void)out_size; (void)ws_size;
    const float* x     = (const float*)d_in[0];
    const float* Wenc  = (const float*)d_in[1];
    const float* benc  = (const float*)d_in[2];
    const float* Wg    = (const float*)d_in[3];
    const float* U     = (const float*)d_in[4];
    const float* V     = (const float*)d_in[5];
    const float* gamma = (const float*)d_in[6];
    float* out = (float*)d_out;

    char* ws = (char*)d_ws;
    size_t o = 0;
    auto alloc = [&](size_t bytes) -> char* {
        char* p = ws + o;
        o = (o + bytes + 255) & ~(size_t)255;
        return p;
    };
    bf16_t* xb   = (bf16_t*)alloc((size_t)NTOK * DM * 2);
    bf16_t* wb   = (bf16_t*)alloc((size_t)DM * DM * 2);
    bf16_t* wgb  = (bf16_t*)alloc((size_t)NEXP * DM * 2);
    bf16_t* ub   = (bf16_t*)alloc((size_t)NEXP * RLOW * DM * 2);
    bf16_t* vbw  = (bf16_t*)alloc((size_t)NEXP * DM * RLOW * 2);
    bf16_t* encb = (bf16_t*)alloc((size_t)NTOK * DM * 2);
    float*  wts  = (float*)alloc((size_t)NTOK * 2 * 4);
    int*    toke = (int*)alloc((size_t)NTOK * 2 * 4);
    int*    counts = (int*)alloc(NEXP * 4);
    int*    offs   = (int*)alloc((NEXP + 1) * 4);
    int*    curs   = (int*)alloc(NEXP * 4);
    int*    perm   = (int*)alloc((size_t)NTOK * 2 * 4);
    bf16_t* sg     = (bf16_t*)alloc((size_t)NTOK * 2 * RLOW * 2);

    auto cvt = [&](const float* in, bf16_t* outp, size_t n) {
        size_t n4 = n / 4;
        int blocks = (int)((n4 + 256 * 4 - 1) / (256 * 4));
        if (blocks > 4096) blocks = 4096;
        if (blocks < 1) blocks = 1;
        cvt_f32_bf16<<<blocks, 256, 0, stream>>>(in, outp, n4);
    };
    cvt(x, xb, (size_t)NTOK * DM);
    cvt(Wenc, wb, (size_t)DM * DM);
    cvt(Wg, wgb, (size_t)NEXP * DM);
    cvt(U, ub, (size_t)NEXP * RLOW * DM);
    cvt(V, vbw, (size_t)NEXP * DM * RLOW);

    zero_counts<<<1, 32, 0, stream>>>(counts, curs);
    gemm_encode<<<dim3(NTOK / 128, DM / 128), 256, 0, stream>>>(xb, wb, benc, out, encb);
    router_kernel<<<NTOK / 128, 256, 0, stream>>>(encb, wgb, gamma, wts, toke, counts);
    scan_offsets<<<1, 32, 0, stream>>>(counts, offs);
    fill_perm<<<(NTOK + 255) / 256, 256, 0, stream>>>(toke, offs, curs, perm);
    gemm_expert_s<<<dim3(NTOK / 128, NEXP), 256, 0, stream>>>(encb, ub, wts, perm, offs, counts, sg);
    gemm_expert_delta<<<dim3(NTOK / 128, DM / 128, NEXP), 256, 0, stream>>>(sg, vbw, perm, offs, counts, out);
}
